// LSTMWebNet_4767413699187
// MI455X (gfx1250) — matrix operand order repaired and checked by reading, not compiled, not run on hardware
//
#include <hip/hip_runtime.h>
#include <hip/hip_bf16.h>
#include <stdint.h>

typedef __attribute__((ext_vector_type(16))) __bf16 v16bf;
typedef __attribute__((ext_vector_type(8)))  __bf16 bf16x8;
typedef __attribute__((ext_vector_type(8)))  float  v8f;
typedef __attribute__((ext_vector_type(4)))  float  f32x4;
typedef __attribute__((ext_vector_type(4)))  int    i32x4;

#define AS1 __attribute__((address_space(1)))
#define AS3 __attribute__((address_space(3)))

#define KS  32   // K per LDS stage (one WMMA k-step)
#define KSP 40   // padded LDS row stride (bf16) -> conflict-free banks
#define KC  1024 // K per block (K-split chunk)

#if defined(__HIP_DEVICE_COMPILE__) && defined(__has_builtin)
#  if __has_builtin(__builtin_amdgcn_global_load_async_to_lds_b128)
#    define HAVE_ASYNC_LDS 1
#  endif
#endif

__device__ __forceinline__ const AS1 float* g1f(const float* p) {
    return (const AS1 float*)(uintptr_t)p;
}
__device__ __forceinline__ const AS1 __bf16* g1b(const __bf16* p) {
    return (const AS1 __bf16*)(uintptr_t)p;
}
__device__ __forceinline__ AS1 float* g1fw(float* p) {
    return (AS1 float*)(uintptr_t)p;
}
__device__ __forceinline__ AS3 i32x4* l3v(void* p) {
    // generic LDS pointer: low 32 bits are the LDS offset (ISA aperture rule)
    return (AS3 i32x4*)(uint32_t)(uintptr_t)p;
}

__device__ __forceinline__ float sigmoidf_(float x) {
    return 1.0f / (1.0f + __expf(-x));
}

__device__ __forceinline__ void cvt8(v16bf& d, int o, f32x4 a, f32x4 b) {
    d[o + 0] = (__bf16)a.x; d[o + 1] = (__bf16)a.y;
    d[o + 2] = (__bf16)a.z; d[o + 3] = (__bf16)a.w;
    d[o + 4] = (__bf16)b.x; d[o + 5] = (__bf16)b.y;
    d[o + 6] = (__bf16)b.z; d[o + 7] = (__bf16)b.w;
}

// ======================= pass 0: f32 -> bf16 X pre-convert =======================
__global__ __launch_bounds__(256)
void cvt_f32_to_bf16(const float* __restrict__ src, __bf16* __restrict__ dst, int n8) {
    int t = blockIdx.x * blockDim.x + threadIdx.x;
    if (t >= n8) return;
    const AS1 f32x4* s = (const AS1 f32x4*)(g1f(src)) + (size_t)t * 2;
    f32x4 a = __builtin_nontemporal_load(s);
    f32x4 b = __builtin_nontemporal_load(s + 1);
    bf16x8 o;
    o[0] = (__bf16)a.x; o[1] = (__bf16)a.y; o[2] = (__bf16)a.z; o[3] = (__bf16)a.w;
    o[4] = (__bf16)b.x; o[5] = (__bf16)b.y; o[6] = (__bf16)b.z; o[7] = (__bf16)b.w;
    AS1 bf16x8* d = (AS1 bf16x8*)(uintptr_t)(dst) + t;
    __builtin_nontemporal_store(o, d);
}

// ======================= pass 1: K-split partial GEMM, LDS-staged B ==============
__device__ __forceinline__ void fill_stage(__bf16* buf, const __bf16* Xb, int K,
                                           int kglob, int tid) {
    // 256 rows x 64B per stage; 256 threads x 4 transfers of 16B, coalesced
#pragma unroll
    for (int rep = 0; rep < 4; ++rep) {
        int flat = rep * 256 + tid;
        int b = flat >> 2, seg = flat & 3;
        const AS1 __bf16* g = g1b(Xb) + (size_t)b * K + kglob + seg * 8;
        __bf16* l = buf + b * KSP + seg * 8;
#ifdef HAVE_ASYNC_LDS
        __builtin_amdgcn_global_load_async_to_lds_b128(
            (AS1 i32x4*)(uintptr_t)g, l3v(l), 0, 0);
#else
        bf16x8 v = *(const AS1 bf16x8*)g;
        *(bf16x8*)l = v;
#endif
    }
}

__device__ __forceinline__ void wait_fill() {
#ifdef HAVE_ASYNC_LDS
#  if __has_builtin(__builtin_amdgcn_s_wait_asynccnt)
    __builtin_amdgcn_s_wait_asynccnt(0);
#  else
    asm volatile("s_wait_asynccnt 0x0" ::: "memory");
#  endif
#endif
}

__device__ __forceinline__ v16bf loadB(const __bf16* cur, int b, int half) {
    const bf16x8* lp = (const bf16x8*)(cur + b * KSP + half * 16);
    return __builtin_shufflevector(lp[0], lp[1], 0, 1, 2, 3, 4, 5, 6, 7,
                                   8, 9, 10, 11, 12, 13, 14, 15);
}

// Block = 8 waves x 16 M-rows = 128 rows, full 256 batch, K range [by*KC, +KC).
// B chunk staged in LDS (shared by 8 waves); W streamed NT, read once device-wide.
__global__ __launch_bounds__(256)
void gemm_partial(const float* __restrict__ W, const __bf16* __restrict__ Xb,
                  int K, int M, float* __restrict__ outP)
{
    extern __shared__ __bf16 lds[];          // 2 * 256*KSP bf16 = 40960 B
    __bf16* buf0 = lds;
    __bf16* buf1 = lds + 256 * KSP;

    const int tid  = threadIdx.x;
    const int lane = tid & 31;
    const int wave = tid >> 5;
    const int m0   = (blockIdx.x * 8 + wave) * 16;
    const int r    = lane & 15;
    const int half = lane >> 4;
    const int koff = blockIdx.y * KC;
    float* outPc = outP + (size_t)blockIdx.y * ((size_t)M * 256);

    v8f acc[16];
#pragma unroll
    for (int t = 0; t < 16; ++t)
        acc[t] = (v8f){0.f, 0.f, 0.f, 0.f, 0.f, 0.f, 0.f, 0.f};

    const AS1 float* wrow = g1f(W) + (size_t)(m0 + r) * K + koff + half * 16;

    fill_stage(buf0, Xb, K, koff, tid);
    wait_fill();
    __syncthreads();

#pragma unroll 1
    for (int s = 0; s < KC / KS; ++s) {
        __bf16* cur = (s & 1) ? buf1 : buf0;
        __bf16* nxt = (s & 1) ? buf0 : buf1;
        if (s + 1 < KC / KS)
            fill_stage(nxt, Xb, K, koff + (s + 1) * KS, tid);

        // A tile 16x32 bf16 from W (NT stream)
        const AS1 f32x4* ap = (const AS1 f32x4*)(wrow + s * KS);
        f32x4 a0 = __builtin_nontemporal_load(ap);
        f32x4 a1 = __builtin_nontemporal_load(ap + 1);
        f32x4 a2 = __builtin_nontemporal_load(ap + 2);  // +8 floats
        f32x4 a3 = __builtin_nontemporal_load(ap + 3);
        v16bf A;
        cvt8(A, 0, a0, a1);
        cvt8(A, 8, a2, a3);

        // software-pipelined B fragments: next tile's ds_loads behind current wmma
        v16bf Bnext = loadB(cur, r, half);
#pragma unroll
        for (int t = 0; t < 16; ++t) {
            v16bf Bcur = Bnext;
            if (t < 15)
                Bnext = loadB(cur, (t + 1) * 16 + r, half);
            acc[t] = __builtin_amdgcn_wmma_f32_16x16x32_bf16(
                false, A, false, Bcur, (short)0, acc[t], false, false);
        }

        wait_fill();
        __syncthreads();
    }

    // partial out, layout [b][m] (coalesced for pass 2), NT stores
#pragma unroll
    for (int t = 0; t < 16; ++t) {
        const int b = t * 16 + r;
        AS1 float* pd = g1fw(outPc) + (size_t)b * M + m0 + 8 * half;
        f32x4 lo = {acc[t][0], acc[t][1], acc[t][2], acc[t][3]};
        f32x4 hi = {acc[t][4], acc[t][5], acc[t][6], acc[t][7]};
        __builtin_nontemporal_store(lo, (AS1 f32x4*)pd);
        __builtin_nontemporal_store(hi, (AS1 f32x4*)pd + 1);
    }
}

// ======================= pass 2: 6-way reduce + fused LSTM epilogue ==============
__global__ __launch_bounds__(256)
void reduce_lstm(const float* __restrict__ P, int M,
                 const float* __restrict__ Hin, const float* __restrict__ Cin,
                 const float* __restrict__ Wih, const float* __restrict__ Whh,
                 const float* __restrict__ bih, const float* __restrict__ bhh,
                 const float* __restrict__ Whr,
                 float* __restrict__ outNew, float* __restrict__ outCells)
{
    const size_t tid = (size_t)blockIdx.x * 256 + threadIdx.x;  // = b*M + m
    const size_t NMB = (size_t)M * 256;

    float x = 0.f;
#pragma unroll
    for (int p = 0; p < 6; ++p)
        x += __builtin_nontemporal_load(g1f(P) + (size_t)p * NMB + tid);
    x = fmaxf(x, 0.0f);                                  // in_activation ReLU

    float wih[16], whh[16], bias[16], whr[4];
#pragma unroll
    for (int g = 0; g < 16; ++g) {
        wih[g]  = Wih[g];
        whh[g]  = Whh[g];
        bias[g] = bih[g] + bhh[g];
    }
#pragma unroll
    for (int q = 0; q < 4; ++q) whr[q] = Whr[q];

    float h0 = __builtin_nontemporal_load(g1f(Hin) + tid);
    f32x4 c0 = __builtin_nontemporal_load((const AS1 f32x4*)(g1f(Cin) + tid * 4));
    float c0a[4] = {c0.x, c0.y, c0.z, c0.w};

    float ct[4];
    float hp = 0.0f;
#pragma unroll
    for (int q = 0; q < 4; ++q) {                        // gate order i,f,g,o
        float gi = fmaf(x, wih[q],      fmaf(h0, whh[q],      bias[q]));
        float gf = fmaf(x, wih[4 + q],  fmaf(h0, whh[4 + q],  bias[4 + q]));
        float gg = fmaf(x, wih[8 + q],  fmaf(h0, whh[8 + q],  bias[8 + q]));
        float go = fmaf(x, wih[12 + q], fmaf(h0, whh[12 + q], bias[12 + q]));
        float c1 = sigmoidf_(gf) * c0a[q] + sigmoidf_(gi) * tanhf(gg);
        ct[q] = c1;
        hp = fmaf(sigmoidf_(go) * tanhf(c1), whr[q], hp);
    }

    __builtin_nontemporal_store(fmaxf(hp, 0.0f), g1fw(outNew) + tid);
    f32x4 co = {ct[0], ct[1], ct[2], ct[3]};
    __builtin_nontemporal_store(co, (AS1 f32x4*)(g1fw(outCells) + tid * 4));
}

// ======================= fallback: fused register-only kernel ====================
template <bool XBF16>
__device__ __forceinline__ void gemm_segment(v8f acc[16], const float* W,
                                             const float* Xf, const __bf16* Xb,
                                             int K, int m0, int r, int half)
{
    const AS1 float* wrow = g1f(W) + (size_t)(m0 + r) * K;
#pragma unroll 1
    for (int k0 = 0; k0 < K; k0 += 32) {
        const AS1 f32x4* ap = (const AS1 f32x4*)(wrow + k0 + half * 16);
        f32x4 a0 = __builtin_nontemporal_load(ap);
        f32x4 a1 = __builtin_nontemporal_load(ap + 1);
        f32x4 a2 = __builtin_nontemporal_load(ap + 2);
        f32x4 a3 = __builtin_nontemporal_load(ap + 3);
        v16bf A;
        cvt8(A, 0, a0, a1);
        cvt8(A, 8, a2, a3);
#pragma unroll
        for (int t = 0; t < 16; ++t) {
            const int b = t * 16 + r;
            v16bf Bm;
            if (XBF16) {
                Bm = *(const AS1 v16bf*)(g1b(Xb) + (size_t)b * K + k0 + half * 16);
            } else {
                const AS1 f32x4* bp =
                    (const AS1 f32x4*)(g1f(Xf) + (size_t)b * K + k0 + half * 16);
                f32x4 b0 = bp[0], b1 = bp[1], b2 = bp[2], b3 = bp[3];
                cvt8(Bm, 0, b0, b1);
                cvt8(Bm, 8, b2, b3);
            }
            acc[t] = __builtin_amdgcn_wmma_f32_16x16x32_bf16(
                false, A, false, Bm, (short)0, acc[t], false, false);
        }
    }
}

template <bool XBF16>
__global__ __launch_bounds__(256)
void lstmweb_gemm_lstm(const float* __restrict__ W1, const float* __restrict__ W2,
                       const float* __restrict__ X1f, const float* __restrict__ X2f,
                       const __bf16* __restrict__ X1b, const __bf16* __restrict__ X2b,
                       int K1, int K2, int M,
                       const float* __restrict__ Hin, const float* __restrict__ Cin,
                       const float* __restrict__ Wih, const float* __restrict__ Whh,
                       const float* __restrict__ bih, const float* __restrict__ bhh,
                       const float* __restrict__ Whr,
                       float* __restrict__ outNew, float* __restrict__ outCells)
{
    const int lane = threadIdx.x & 31;
    const int wave = threadIdx.x >> 5;
    const int m0   = (blockIdx.x * 8 + wave) * 16;
    const int r    = lane & 15;
    const int half = lane >> 4;

    v8f acc[16];
#pragma unroll
    for (int t = 0; t < 16; ++t)
        acc[t] = (v8f){0.f, 0.f, 0.f, 0.f, 0.f, 0.f, 0.f, 0.f};

    gemm_segment<XBF16>(acc, W1, X1f, X1b, K1, m0, r, half);
    gemm_segment<XBF16>(acc, W2, X2f, X2b, K2, m0, r, half);

    float wih[16], whh[16], bias[16], whr[4];
#pragma unroll
    for (int g = 0; g < 16; ++g) {
        wih[g]  = Wih[g];
        whh[g]  = Whh[g];
        bias[g] = bih[g] + bhh[g];
    }
#pragma unroll
    for (int q = 0; q < 4; ++q) whr[q] = Whr[q];

#pragma unroll 1
    for (int t = 0; t < 16; ++t) {
        const int b = t * 16 + r;
#pragma unroll
        for (int j = 0; j < 8; ++j) {
            const int m = m0 + j + 8 * half;
            const size_t idx = (size_t)b * M + m;
            float x  = fmaxf(acc[t][j], 0.0f);
            float h0 = __builtin_nontemporal_load(g1f(Hin) + idx);
            f32x4 c0 = __builtin_nontemporal_load(
                (const AS1 f32x4*)(g1f(Cin) + idx * 4));
            float c0a[4] = {c0.x, c0.y, c0.z, c0.w};
            float ct[4];
            float hp = 0.0f;
#pragma unroll
            for (int q = 0; q < 4; ++q) {
                float gi = fmaf(x, wih[q],      fmaf(h0, whh[q],      bias[q]));
                float gf = fmaf(x, wih[4 + q],  fmaf(h0, whh[4 + q],  bias[4 + q]));
                float gg = fmaf(x, wih[8 + q],  fmaf(h0, whh[8 + q],  bias[8 + q]));
                float go = fmaf(x, wih[12 + q], fmaf(h0, whh[12 + q], bias[12 + q]));
                float c1 = sigmoidf_(gf) * c0a[q] + sigmoidf_(gi) * tanhf(gg);
                ct[q] = c1;
                hp = fmaf(sigmoidf_(go) * tanhf(c1), whr[q], hp);
            }
            __builtin_nontemporal_store(fmaxf(hp, 0.0f), g1fw(outNew) + idx);
            f32x4 co = {ct[0], ct[1], ct[2], ct[3]};
            __builtin_nontemporal_store(co, (AS1 f32x4*)(g1fw(outCells) + idx * 4));
        }
    }
}

// =================================================================================
extern "C" void kernel_launch(void* const* d_in, const int* in_sizes, int n_in,
                              void* d_out, int out_size, void* d_ws, size_t ws_size,
                              hipStream_t stream) {
    const float* i_in   = (const float*)d_in[0];
    const float* h_in   = (const float*)d_in[1];
    const float* o_in   = (const float*)d_in[2];
    const float* hcells = (const float*)d_in[3];
    const float* ocells = (const float*)d_in[4];
    const float* i2h    = (const float*)d_in[5];
    const float* h2h    = (const float*)d_in[6];
    const float* i2o    = (const float*)d_in[7];
    const float* h2o    = (const float*)d_in[8];
    const float* hWih   = (const float*)d_in[9];
    const float* hWhh   = (const float*)d_in[10];
    const float* hbih   = (const float*)d_in[11];
    const float* hbhh   = (const float*)d_in[12];
    const float* hWhr   = (const float*)d_in[13];
    const float* oWih   = (const float*)d_in[14];
    const float* oWhh   = (const float*)d_in[15];
    const float* obih   = (const float*)d_in[16];
    const float* obhh   = (const float*)d_in[17];
    const float* oWhr   = (const float*)d_in[18];

    const int B_ = 256, NI = 2048, NH = 4096, NO = 512;

    float* out = (float*)d_out;
    float* outNewH = out;
    float* outNewO = out + (size_t)B_ * NH;
    float* outHC   = outNewO + (size_t)B_ * NO;
    float* outOC   = outHC + (size_t)B_ * NH * 4;

    const size_t nXi = (size_t)B_ * NI;
    const size_t nXh = (size_t)B_ * NH;
    const size_t bytesX  = (nXi + nXh) * 2;
    const size_t pHf     = 6ull * B_ * NH;
    const size_t pOf     = 6ull * B_ * NO;
    const size_t needAll = bytesX + (pHf + pOf) * 4;

    __bf16* Xi_b = (__bf16*)d_ws;
    __bf16* Xh_b = Xi_b + nXi;
    float*  pH   = (float*)((char*)d_ws + bytesX);
    float*  pO   = pH + pHf;

    dim3 block(256);
    const size_t ldsBytes = 2 * 256 * KSP * sizeof(__bf16);  // 40960

    if (ws_size >= needAll) {
        cvt_f32_to_bf16<<<dim3((unsigned)(nXi / 8 / 256)), block, 0, stream>>>(
            i_in, Xi_b, (int)(nXi / 8));
        cvt_f32_to_bf16<<<dim3((unsigned)(nXh / 8 / 256)), block, 0, stream>>>(
            h_in, Xh_b, (int)(nXh / 8));

        gemm_partial<<<dim3(NH / 128, 2), block, ldsBytes, stream>>>(
            i2h, Xi_b, NI, NH, pH);
        gemm_partial<<<dim3(NH / 128, 4), block, ldsBytes, stream>>>(
            h2h, Xh_b, NH, NH, pH + 2ull * B_ * NH);
        gemm_partial<<<dim3(NO / 128, 2), block, ldsBytes, stream>>>(
            i2o, Xi_b, NI, NO, pO);
        gemm_partial<<<dim3(NO / 128, 4), block, ldsBytes, stream>>>(
            h2o, Xh_b, NH, NO, pO + 2ull * B_ * NO);

        reduce_lstm<<<dim3(NH), block, 0, stream>>>(
            pH, NH, h_in, hcells, hWih, hWhh, hbih, hbhh, hWhr, outNewH, outHC);
        reduce_lstm<<<dim3(NO), block, 0, stream>>>(
            pO, NO, o_in, ocells, oWih, oWhh, obih, obhh, oWhr, outNewO, outOC);
    } else if (ws_size >= bytesX) {
        cvt_f32_to_bf16<<<dim3((unsigned)(nXi / 8 / 256)), block, 0, stream>>>(
            i_in, Xi_b, (int)(nXi / 8));
        cvt_f32_to_bf16<<<dim3((unsigned)(nXh / 8 / 256)), block, 0, stream>>>(
            h_in, Xh_b, (int)(nXh / 8));

        lstmweb_gemm_lstm<true><<<dim3(NH / 128), block, 0, stream>>>(
            i2h, h2h, i_in, h_in, Xi_b, Xh_b, NI, NH, NH,
            h_in, hcells, hWih, hWhh, hbih, hbhh, hWhr, outNewH, outHC);
        lstmweb_gemm_lstm<true><<<dim3(NO / 128), block, 0, stream>>>(
            i2o, h2o, i_in, h_in, Xi_b, Xh_b, NI, NH, NO,
            o_in, ocells, oWih, oWhh, obih, obhh, oWhr, outNewO, outOC);
    } else {
        lstmweb_gemm_lstm<false><<<dim3(NH / 128), block, 0, stream>>>(
            i2h, h2h, i_in, h_in, nullptr, nullptr, NI, NH, NH,
            h_in, hcells, hWih, hWhh, hbih, hbhh, hWhr, outNewH, outHC);
        lstmweb_gemm_lstm<false><<<dim3(NO / 128), block, 0, stream>>>(
            i2o, h2o, i_in, h_in, nullptr, nullptr, NI, NH, NO,
            o_in, ocells, oWih, oWhh, obih, obhh, oWhr, outNewO, outOC);
    }
}
